// SSConv2d_62483184222746
// MI455X (gfx1250) — matrix operand order repaired and checked by reading, not compiled, not run on hardware
//
#include <hip/hip_runtime.h>
#include <hip/hip_bf16.h>

// SSConv2d on MI455X (gfx1250): routed 3x3 conv as dense-K implicit GEMM
// (K = 9 taps * 4 groups * 16 cin = 576) using v_wmma_f32_16x16x32_bf16 with
// hi/lo bf16 split of both operands (bf16x3: Ah*Bh + Ah*Bl + Al*Bh) for
// ~fp32 accuracy. Weights pre-reordered into the WMMA A-fragment lane layout
// by a prep kernel; main kernel stages A in LDS in two K-phases (104 KB LDS
// -> 3 WGs/WGP) using GLOBAL_LOAD_ASYNC_TO_LDS_B128, and each wave computes a
// 64oc x 32pix tile (2 row-strips) so every A fragment feeds 6 WMMAs.

typedef __attribute__((ext_vector_type(16))) __bf16 v16bf;
typedef __attribute__((ext_vector_type(8)))  float  v8f;

#define Hdim 256
#define Wdim 256
#define Bdim 8
#define CIN 16
#define OC 64
#define NTAPS 9
#define NCHUNK 18               // 576 / 32
#define A_ELEMS (OC * 576)      // 36864 bf16 per (hi|lo) array
#define PH0_CHUNKS 10           // taps 0..4
#define PH1_CHUNKS 8            // taps 5..8
#define PH0_ELEMS (PH0_CHUNKS * 4 * 32 * 16)   // 20480
#define PH1_ELEMS (PH1_CHUNKS * 4 * 32 * 16)   // 16384
#define HALO 324                // 18 x 18 pixels

#define SS_USE_ASYNC 1

union FragB { unsigned int u[8]; uint4 q[2]; v16bf v; };
union FragA { uint4 q[2]; v16bf v; };

__device__ __forceinline__ unsigned short f32_to_bf16_rne(float f) {
  unsigned int u = __float_as_uint(f);
  unsigned int r = u + 0x7FFFu + ((u >> 16) & 1u);
  return (unsigned short)(r >> 16);
}
__device__ __forceinline__ float bf16_to_f32(unsigned short h) {
  return __uint_as_float(((unsigned int)h) << 16);
}

// 16B global->LDS copy. Async path: VGLOBAL GLOBAL_LOAD_ASYNC_TO_LDS_B128,
// GV mode (vdst = LDS byte offset, vaddr = 64-bit global address), tracked
// with ASYNCcnt; fenced with s_wait_asynccnt before the block barrier.
__device__ __forceinline__ void copy16_g2l(const unsigned short* __restrict__ g,
                                           unsigned short* l) {
#if SS_USE_ASYNC
  unsigned int lds_off = (unsigned int)(uintptr_t)l;
  asm volatile("global_load_async_to_lds_b128 %0, %1, off"
               :: "v"(lds_off), "v"(g) : "memory");
#else
  *(uint4*)l = *(const uint4*)g;
#endif
}
__device__ __forceinline__ void wait_async_copies() {
#if SS_USE_ASYNC
  asm volatile("s_wait_asynccnt 0x0" ::: "memory");
#endif
}

// ---------------------------------------------------------------------------
// Prep: weight (64,64,3,3) f32 -> A-fragment layout [chunk][tile][lane][slot],
// split into bf16 hi + lo. Contiguous k order (k = 16*half + slot, same as B):
// lane = m + 16*(k>>4), slot = k & 15.
// ---------------------------------------------------------------------------
__global__ void ssconv_prep_weights(const float* __restrict__ w,
                                    unsigned short* __restrict__ Ah,
                                    unsigned short* __restrict__ Al) {
  int i = blockIdx.x * 256 + threadIdx.x;       // 0 .. 36863
  if (i >= A_ELEMS) return;
  int tap  = i % 9;
  int rest = i / 9;
  int ic   = rest % 64;
  int oc   = rest / 64;
  float wv = w[i];                              // flat OIHW == (oc*64+ic)*9 + tap

  int K     = tap * 64 + ic;
  int chunk = K >> 5;
  int kl    = K & 31;
  int tile  = oc >> 4;
  int m     = oc & 15;
  int lane  = m + 16 * (kl >> 4);
  int slot  = kl & 15;
  int dst   = ((chunk * 4 + tile) * 32 + lane) * 16 + slot;

  unsigned short hi = f32_to_bf16_rne(wv);
  unsigned short lo = f32_to_bf16_rne(wv - bf16_to_f32(hi));
  Ah[dst] = hi;
  Al[dst] = lo;
}

// ---------------------------------------------------------------------------
// Main: block = 8 waves, tile = 16(W) x 16(H) output pixels; each wave owns
// two adjacent row-strips (N = 32) and all 4 oc-tiles (= the 4 out-groups).
// ---------------------------------------------------------------------------
__global__ __launch_bounds__(256) void ssconv_main(
    const float* __restrict__ x, const int* __restrict__ in_index,
    const float* __restrict__ bias,
    const unsigned short* __restrict__ Ahg, const unsigned short* __restrict__ Alg,
    float* __restrict__ sel_out, float* __restrict__ idx_out) {
  __shared__ unsigned short sAh[PH0_ELEMS];     // 40960 B (max of both phases)
  __shared__ unsigned short sAl[PH0_ELEMS];     // 40960 B
  __shared__ unsigned short sxh[HALO * CIN];    // 10368 B
  __shared__ unsigned short sxl[HALO * CIN];    // 10368 B
  __shared__ int            sgrp[HALO];         //  1296 B  => ~104 KB total

  const int tid = threadIdx.x;

  // ---- Phase-0 A fill (chunks 0..9, hi+lo) : 5120 x 16B, async to LDS ----
  {
#pragma unroll
    for (int j = 0; j < PH0_ELEMS / 8 / 256; ++j) {   // 10 iters
      int idx = tid + j * 256;                        // uint4 index
      copy16_g2l(Ahg + idx * 8, sAh + idx * 8);
      copy16_g2l(Alg + idx * 8, sAl + idx * 8);
    }
  }

  const int b  = blockIdx.z;
  const int y0 = blockIdx.y * 16;
  const int x0 = blockIdx.x * 16;

  // ---- Stage 18x18 halo of x (bf16 hi/lo) + in-group, SAME zero pad ----
  for (int hp = tid; hp < HALO; hp += 256) {
    int iy = hp / 18, ix = hp % 18;
    int gy = y0 - 1 + iy, gx = x0 - 1 + ix;
    bool ok = (gy >= 0) && (gy < Hdim) && (gx >= 0) && (gx < Wdim);
    float v[CIN];
    int g = -1;
    if (ok) {
      size_t p = (((size_t)b * Hdim + gy) * Wdim + gx);
      const float4* px = (const float4*)(x + p * CIN);
#pragma unroll
      for (int q = 0; q < 4; ++q) {
        float4 t = px[q];
        v[q * 4 + 0] = t.x; v[q * 4 + 1] = t.y;
        v[q * 4 + 2] = t.z; v[q * 4 + 3] = t.w;
      }
      g = in_index[p];
    } else {
#pragma unroll
      for (int q = 0; q < CIN; ++q) v[q] = 0.f;
    }
    sgrp[hp] = g;
#pragma unroll
    for (int q = 0; q < CIN; ++q) {
      unsigned short h = f32_to_bf16_rne(v[q]);
      sxh[hp * CIN + q] = h;
      sxl[hp * CIN + q] = f32_to_bf16_rne(v[q] - bf16_to_f32(h));
    }
  }
  wait_async_copies();
  __syncthreads();

  const int wv   = tid >> 5;   // wave id: owns output rows 2*wv, 2*wv+1
  const int lane = tid & 31;
  const int n    = lane & 15;  // output column (N dim of WMMA)
  const int hh   = lane >> 4;  // lane half

  // C/D 16x16 f32 layout: lane n/(n+16), VGPR r -> M = r + 8*hh. Seed w/ bias.
  v8f acc[2][4];
#pragma unroll
  for (int t = 0; t < 4; ++t)
#pragma unroll
    for (int r = 0; r < 8; ++r) {
      float bv = bias[t * 16 + hh * 8 + r];
      acc[0][t][r] = bv;
      acc[1][t][r] = bv;
    }

#pragma unroll
  for (int phase = 0; phase < 2; ++phase) {
    if (phase == 1) {
      // Re-fill A buffer with chunks 10..17 (taps 5..8).
      __syncthreads();          // all waves done reading phase-0 A
#pragma unroll
      for (int j = 0; j < PH1_ELEMS / 8 / 256; ++j) {  // 8 iters
        int idx = tid + j * 256;
        copy16_g2l(Ahg + PH0_ELEMS + idx * 8, sAh + idx * 8);
        copy16_g2l(Alg + PH0_ELEMS + idx * 8, sAl + idx * 8);
      }
      wait_async_copies();
      __syncthreads();
    }
    const int tap_lo = (phase == 0) ? 0 : 5;
    const int tap_hi = (phase == 0) ? 5 : 9;
    const int chunk0 = (phase == 0) ? 0 : 10;

#pragma unroll
    for (int tap = tap_lo; tap < tap_hi; ++tap) {
      const int dy = tap / 3 - 1, dx = tap % 3 - 1;
      const int ix  = n + 1 + dx;
      const int py0 = (2 * wv + 1 + dy) * 18 + ix;    // halo pixel, row strip 0
      const int py1 = py0 + 18;                       // row strip 1
      const int pg0 = sgrp[py0];
      const int pg1 = sgrp[py1];
      FragB rawh0, rawl0, rawh1, rawl1;
      {
        const uint4* p0h = (const uint4*)(sxh + py0 * CIN);
        const uint4* p0l = (const uint4*)(sxl + py0 * CIN);
        const uint4* p1h = (const uint4*)(sxh + py1 * CIN);
        const uint4* p1l = (const uint4*)(sxl + py1 * CIN);
        rawh0.q[0] = p0h[0]; rawh0.q[1] = p0h[1];
        rawl0.q[0] = p0l[0]; rawl0.q[1] = p0l[1];
        rawh1.q[0] = p1h[0]; rawh1.q[1] = p1h[1];
        rawl1.q[0] = p1l[0]; rawl1.q[1] = p1l[1];
      }
#pragma unroll
      for (int halfsel = 0; halfsel < 2; ++halfsel) {
        const int cl = tap * 2 + halfsel - chunk0;    // chunk index in LDS
        // B 32x16 layout: lane half hh = K half; this lane's 16 slots are
        // cin 0..15 of group (2*halfsel + hh). Mask by routing match.
        const int gsel = halfsel * 2 + hh;
        const bool m0 = (pg0 == gsel);
        const bool m1 = (pg1 == gsel);
        FragB Bh0, Bl0, Bh1, Bl1;
#pragma unroll
        for (int j = 0; j < 8; ++j) {
          Bh0.u[j] = m0 ? rawh0.u[j] : 0u;
          Bl0.u[j] = m0 ? rawl0.u[j] : 0u;
          Bh1.u[j] = m1 ? rawh1.u[j] : 0u;
          Bl1.u[j] = m1 ? rawl1.u[j] : 0u;
        }
        // Load all 4 oc-tiles' A fragments (hi+lo) into distinct registers,
        // then issue 24 WMMAs over 8 independent accumulator chains.
        FragA AhT[4], AlT[4];
#pragma unroll
        for (int t = 0; t < 4; ++t) {
          const uint4* pa = (const uint4*)(sAh + ((cl * 4 + t) * 32 + lane) * 16);
          const uint4* pb = (const uint4*)(sAl + ((cl * 4 + t) * 32 + lane) * 16);
          AhT[t].q[0] = pa[0]; AhT[t].q[1] = pa[1];
          AlT[t].q[0] = pb[0]; AlT[t].q[1] = pb[1];
        }
#pragma unroll
        for (int t = 0; t < 4; ++t) {
          acc[0][t] = __builtin_amdgcn_wmma_f32_16x16x32_bf16(
              false, AhT[t].v, false, Bh0.v, (short)0, acc[0][t], false, false);
          acc[1][t] = __builtin_amdgcn_wmma_f32_16x16x32_bf16(
              false, AhT[t].v, false, Bh1.v, (short)0, acc[1][t], false, false);
          acc[0][t] = __builtin_amdgcn_wmma_f32_16x16x32_bf16(
              false, AhT[t].v, false, Bl0.v, (short)0, acc[0][t], false, false);
          acc[1][t] = __builtin_amdgcn_wmma_f32_16x16x32_bf16(
              false, AhT[t].v, false, Bl1.v, (short)0, acc[1][t], false, false);
          acc[0][t] = __builtin_amdgcn_wmma_f32_16x16x32_bf16(
              false, AlT[t].v, false, Bh0.v, (short)0, acc[0][t], false, false);
          acc[1][t] = __builtin_amdgcn_wmma_f32_16x16x32_bf16(
              false, AlT[t].v, false, Bh1.v, (short)0, acc[1][t], false, false);
        }
      }
    }
  }

  // ---- Maxout epilogue per row strip: oc-tile t == out-group t ----
#pragma unroll
  for (int row = 0; row < 2; ++row) {
    float sc[4];
#pragma unroll
    for (int t = 0; t < 4; ++t) {
      float s = acc[row][t][0];
#pragma unroll
      for (int r = 1; r < 8; ++r) s = fmaxf(s, acc[row][t][r]);
      float o = __shfl_xor(s, 16, 32);     // combine M 0-7 with M 8-15 halves
      sc[t] = fmaxf(s, o);
    }
    int widx = 0;
    float best = sc[0];
#pragma unroll
    for (int t = 1; t < 4; ++t) {
      if (sc[t] > best) { best = sc[t]; widx = t; }  // first-max (jnp.argmax)
    }
    v8f selv = acc[row][0];
#pragma unroll
    for (int t = 1; t < 4; ++t) {
      bool take = (widx == t);
#pragma unroll
      for (int r = 0; r < 8; ++r) selv[r] = take ? acc[row][t][r] : selv[r];
    }
    const int oy = y0 + 2 * wv + row, ox = x0 + n;
    const size_t p = (((size_t)b * Hdim + oy) * Wdim + ox);
    float* outp = sel_out + p * 16 + hh * 8;         // couts 8*hh .. 8*hh+7
    ((float4*)outp)[0] = make_float4(selv[0], selv[1], selv[2], selv[3]);
    ((float4*)outp)[1] = make_float4(selv[4], selv[5], selv[6], selv[7]);
    if (hh == 0) idx_out[p] = (float)widx;
  }
}

// ---------------------------------------------------------------------------
extern "C" void kernel_launch(void* const* d_in, const int* in_sizes, int n_in,
                              void* d_out, int out_size, void* d_ws, size_t ws_size,
                              hipStream_t stream) {
  const float* x        = (const float*)d_in[0];
  const int*   in_index = (const int*)d_in[1];
  const float* weight   = (const float*)d_in[2];
  const float* bias     = (const float*)d_in[3];

  unsigned short* Ah = (unsigned short*)d_ws;          // 73728 B
  unsigned short* Al = Ah + A_ELEMS;                   // 73728 B (ws >= 147456 B)

  float* sel = (float*)d_out;
  float* idx = sel + (size_t)Bdim * Hdim * Wdim * CIN; // out_index stored as f32

  ssconv_prep_weights<<<(A_ELEMS + 255) / 256, 256, 0, stream>>>(weight, Ah, Al);

  dim3 grid(Wdim / 16, Hdim / 16, Bdim);               // (16, 16, 8)
  ssconv_main<<<grid, 256, 0, stream>>>(x, in_index, bias, Ah, Al, sel, idx);
}
